// GCNGRU_Single_58514634440853
// MI455X (gfx1250) — matrix operand order repaired and checked by reading, not compiled, not run on hardware
//
#include <hip/hip_runtime.h>
#include <hip/hip_bf16.h>
#include <cstdint>
#include <cstddef>

// Problem constants (reference: B=128, W=64, S=64, F=64, H=128, HOR=12)
#define BD  128   // batch
#define TT  64    // timesteps (= W)
#define HH  128   // hidden
#define G3  384   // 3*H
#define HOR 12

typedef __attribute__((ext_vector_type(16))) _Float16 v16h;
typedef __attribute__((ext_vector_type(8)))  _Float16 v8h;
typedef __attribute__((ext_vector_type(8)))  float    v8f;

static __device__ __forceinline__ v16h cat8(v8h lo, v8h hi) {
  return __builtin_shufflevector(lo, hi, 0,1,2,3,4,5,6,7,8,9,10,11,12,13,14,15);
}

static __device__ __forceinline__ v8f wmma_f16(v16h a, v16h b, v8f c) {
  // v_wmma_f32_16x16x32_f16  (neg_a, A, neg_b, B, c_mod, C, reuse_a, reuse_b)
  return __builtin_amdgcn_wmma_f32_16x16x32_f16(false, a, false, b, (short)0, c,
                                                false, false);
}

// A-operand (16x32 f16 tile) from a row-major f16 matrix, row stride ld (halves).
// Per ISA layout: lane holds row M=lane%16, K-pairs at {kb+(lane/16)*8 .. +8} and +16.
static __device__ __forceinline__ v16h load_a_f16(const _Float16* M, int ld,
                                                  int row0, int kb, int lane) {
  int r = row0 + (lane & 15);
  int s = kb + ((lane >> 4) << 3);
  v8h lo = *(const v8h*)(M + (size_t)r * ld + s);
  v8h hi = *(const v8h*)(M + (size_t)r * ld + s + 16);
  return cat8(lo, hi);
}

// A-operand built from an fp32 row (convert to f16 on the fly).
static __device__ __forceinline__ v16h load_a_f32row(const float* row, int kb, int lane) {
  int s = kb + ((lane >> 4) << 3);
  v8h lo, hi;
#pragma unroll
  for (int e = 0; e < 8; ++e) {
    lo[e] = (_Float16)row[s + e];
    hi[e] = (_Float16)row[s + 16 + e];
  }
  return cat8(lo, hi);
}

// B-operand (32x16 tile of B), given Bt = B^T stored row-major [N][K] f16.
// Per ISA layout: lane holds column N=lane%16, K at {kb+(lane/16)*8 .. +8} and +16.
static __device__ __forceinline__ v16h load_b_f16(const _Float16* Bt, int ld,
                                                  int n0, int kb, int lane) {
  int r = n0 + (lane & 15);
  int s = kb + ((lane >> 4) << 3);
  v8h lo = *(const v8h*)(Bt + (size_t)r * ld + s);
  v8h hi = *(const v8h*)(Bt + (size_t)r * ld + s + 16);
  return cat8(lo, hi);
}

// ---------------------------------------------------------------------------
// Kernel 1: seq[b,w,:] = (features[b,w,0,:] @ Wr1 + b1) @ Wr2 + b2
// (SAGE layers collapse: agg contribution at s=0 is identically zero.)
// Output directly in xs layout: xs[(w*128 + b)*128 + n]  (f16)
// Grid: 128 blocks x 128 threads (4 waves). Each wave owns one 16-row m-tile.
// ---------------------------------------------------------------------------
__global__ void seq_kernel(const float* __restrict__ feats,
                           const float* __restrict__ Wr1, const float* __restrict__ b1,
                           const float* __restrict__ Wr2, const float* __restrict__ b2,
                           _Float16* __restrict__ xs) {
  extern __shared__ char smem[];
  _Float16* w1t = (_Float16*)smem;                            // [128][64]  Wr1^T
  _Float16* w2t = (_Float16*)(smem + 128 * 64 * 2);           // [128][128] Wr2^T
  _Float16* stg = (_Float16*)(smem + 128 * 64 * 2 + 128 * 128 * 2); // [4][16][128]

  int tid = threadIdx.x;
  for (int i = tid; i < 64 * 128; i += 128) {   // Wr1[k][n] -> w1t[n][k]
    int k = i >> 7, n = i & 127;
    w1t[n * 64 + k] = (_Float16)Wr1[i];
  }
  for (int i = tid; i < 128 * 128; i += 128) {  // Wr2[k][n] -> w2t[n][k]
    int k = i >> 7, n = i & 127;
    w2t[n * 128 + k] = (_Float16)Wr2[i];
  }
  __syncthreads();

  int wave = tid >> 5, lane = tid & 31;
  int m0 = (blockIdx.x * 4 + wave) * 16;        // 512 m-tiles total
  _Float16* st = stg + wave * (16 * 128);
  int col = lane & 15;

  // GEMM1: T1 = feat0 @ Wr1 + b1    (K = 64)
  const float* frow = feats + (size_t)(m0 + (lane & 15)) * 4096; // row stride S*F
  v16h a0 = load_a_f32row(frow, 0, lane);
  v16h a1 = load_a_f32row(frow, 32, lane);
#pragma unroll
  for (int j = 0; j < 8; ++j) {
    v8f c = {};
    c = wmma_f16(a0, load_b_f16(w1t, 64, j * 16, 0, lane), c);
    c = wmma_f16(a1, load_b_f16(w1t, 64, j * 16, 32, lane), c);
    float bias = b1[j * 16 + col];
#pragma unroll
    for (int i = 0; i < 8; ++i) {
      int mr = ((lane >> 4) << 3) + i;
      st[mr * 128 + j * 16 + col] = (_Float16)(c[i] + bias);
    }
  }
  __syncthreads();

  // GEMM2: seq = T1 @ Wr2 + b2      (K = 128)
  v16h A2[4];
#pragma unroll
  for (int kb = 0; kb < 4; ++kb) A2[kb] = load_a_f16(st, 128, 0, kb * 32, lane);
#pragma unroll
  for (int j = 0; j < 8; ++j) {
    v8f c = {};
#pragma unroll
    for (int kb = 0; kb < 4; ++kb)
      c = wmma_f16(A2[kb], load_b_f16(w2t, 128, j * 16, kb * 32, lane), c);
    float bias = b2[j * 16 + col];
#pragma unroll
    for (int i = 0; i < 8; ++i) {
      int m = m0 + ((lane >> 4) << 3) + i;   // m = b*64 + w
      int bb = m >> 6, w = m & 63;
      xs[((size_t)(w * 128 + bb)) * 128 + j * 16 + col] = (_Float16)(c[i] + bias);
    }
  }
}

// ---------------------------------------------------------------------------
// Kernel 2: Gi = X @ Wih^T + bih   (X: [T*128][128] f16, Gi: [T*128][384] f32)
// B^T rows == rows of Wih (row-major), staged to LDS as f16.
// Grid: 512 blocks x 256 threads; block = one 16-row m-tile, wave = 3 n-tiles.
// ---------------------------------------------------------------------------
__global__ void gi_kernel(const _Float16* __restrict__ X,
                          const float* __restrict__ Wih, const float* __restrict__ bih,
                          float* __restrict__ Gi) {
  extern __shared__ char smem[];
  _Float16* wl = (_Float16*)smem;               // [384][128]
  int tid = threadIdx.x;
  for (int i = tid; i < G3 * HH; i += 256) wl[i] = (_Float16)Wih[i];
  __syncthreads();

  int wave = tid >> 5, lane = tid & 31;
  int m0 = blockIdx.x * 16;
  int col = lane & 15;

  v16h A[4];
#pragma unroll
  for (int kb = 0; kb < 4; ++kb) A[kb] = load_a_f16(X, HH, m0, kb * 32, lane);

  for (int q = 0; q < 3; ++q) {
    int n0 = (wave * 3 + q) * 16;
    v8f c = {};
#pragma unroll
    for (int kb = 0; kb < 4; ++kb)
      c = wmma_f16(A[kb], load_b_f16(wl, HH, n0, kb * 32, lane), c);
    float bias = bih[n0 + col];
#pragma unroll
    for (int i = 0; i < 8; ++i) {
      int r = m0 + ((lane >> 4) << 3) + i;
      Gi[(size_t)r * G3 + n0 + col] = c[i] + bias;
    }
  }
}

// ---------------------------------------------------------------------------
// Kernel 3: persistent single-workgroup GRU recurrence (one layer, T steps).
// Whh (f16, 96KB) and h (f16, 32KB) stay resident in LDS for all steps.
// 32 waves; wave w owns m-tile (w>>2) and column tiles {(w&3)*2, (w&3)*2+1},
// computing all three gate slices (r/z/n) for them in registers.
// Gi[t] (h-independent) is preloaded before the WMMA phase so its global
// latency hides under the matrix work.
// ---------------------------------------------------------------------------
__global__ void gru_kernel(const float* __restrict__ Gi,   // [T*128][384]
                           const float* __restrict__ Whh, const float* __restrict__ bhh,
                           _Float16* __restrict__ ys) {    // [T*128][128]
  extern __shared__ char smem[];
  _Float16* wl  = (_Float16*)smem;                 // [384][128] Whh rows (= B^T rows)
  _Float16* hls = (_Float16*)(smem + G3 * HH * 2); // [128][128] hidden state

  int tid = threadIdx.x;                           // 1024 threads = 32 waves
  for (int i = tid; i < G3 * HH; i += 1024) wl[i] = (_Float16)Whh[i];
  for (int i = tid; i < BD * HH; i += 1024) hls[i] = (_Float16)0.0f;
  __syncthreads();

  int wave = tid >> 5, lane = tid & 31;
  int col16 = lane & 15;
  int mt   = wave >> 2;          // both triples of this wave share one m-tile
  int jt0  = (wave & 3) * 2;     // column tiles jt0, jt0+1
  int rsub = ((lane >> 4) << 3); // C-layout row sub-offset (0 or 8)

  // Loop-invariant: recurrent biases for the owned columns.
  float bh[2][3];
#pragma unroll
  for (int p = 0; p < 2; ++p) {
    int col = (jt0 + p) * 16 + col16;
    bh[p][0] = bhh[col];
    bh[p][1] = bhh[128 + col];
    bh[p][2] = bhh[256 + col];
  }

  for (int t = 0; t < TT; ++t) {
    // ---- Preload Gi[t] (independent of h): overlaps the WMMA phase below.
    float gi_r[2][8], gi_z[2][8], gi_n[2][8];
#pragma unroll
    for (int p = 0; p < 2; ++p) {
      int col = (jt0 + p) * 16 + col16;
#pragma unroll
      for (int i = 0; i < 8; ++i) {
        size_t gb = ((size_t)(t * 128 + mt * 16 + rsub + i)) * G3 + col;
        gi_r[p][i] = Gi[gb];
        gi_z[p][i] = Gi[gb + 128];
        gi_n[p][i] = Gi[gb + 256];
      }
    }

    // ---- GEMM phase: gh = h @ Whh^T for the owned tiles (reads h from LDS).
    v16h a[4];
#pragma unroll
    for (int kb = 0; kb < 4; ++kb) a[kb] = load_a_f16(hls, HH, mt * 16, kb * 32, lane);

    v8f cr[2], cz[2], cn[2];
#pragma unroll
    for (int p = 0; p < 2; ++p) {
      int jt = jt0 + p;
      v8f c0 = {}, c1 = {}, c2 = {};
#pragma unroll
      for (int kb = 0; kb < 4; ++kb) {
        c0 = wmma_f16(a[kb], load_b_f16(wl, HH,   0 + jt * 16, kb * 32, lane), c0);
        c1 = wmma_f16(a[kb], load_b_f16(wl, HH, 128 + jt * 16, kb * 32, lane), c1);
        c2 = wmma_f16(a[kb], load_b_f16(wl, HH, 256 + jt * 16, kb * 32, lane), c2);
      }
      cr[p] = c0; cz[p] = c1; cn[p] = c2;
    }
    __syncthreads();   // all h reads complete before any h write

    // ---- Gate math + state update (registers only; writes own h elements).
#pragma unroll
    for (int p = 0; p < 2; ++p) {
      int col = (jt0 + p) * 16 + col16;
#pragma unroll
      for (int i = 0; i < 8; ++i) {
        int m = mt * 16 + rsub + i;
        float hr = cr[p][i] + bh[p][0];
        float hz = cz[p][i] + bh[p][1];
        float hn = cn[p][i] + bh[p][2];
        float rg = 1.0f / (1.0f + __expf(-(gi_r[p][i] + hr)));
        float zg = 1.0f / (1.0f + __expf(-(gi_z[p][i] + hz)));
        float ng = tanhf(gi_n[p][i] + rg * hn);
        float hp = (float)hls[m * HH + col];
        float hN = (1.0f - zg) * ng + zg * hp;
        hls[m * HH + col] = (_Float16)hN;                    // own element only
        ys[((size_t)(t * 128 + m)) * HH + col] = (_Float16)hN;
      }
    }
    __syncthreads();   // h fully updated before next step's GEMM
  }
}

// ---------------------------------------------------------------------------
// Kernel 4: out = h1 @ Wfc + bfc   (128x128 @ 128x12 -> tiny, scalar)
// ---------------------------------------------------------------------------
__global__ void fc_kernel(const _Float16* __restrict__ h1,  // ys1 row block t=63
                          const float* __restrict__ Wfc, const float* __restrict__ bfc,
                          float* __restrict__ out) {
  int tid = blockIdx.x * blockDim.x + threadIdx.x;
  if (tid >= BD * HOR) return;
  int n = tid / HOR, o = tid % HOR;
  float acc = bfc[o];
#pragma unroll 4
  for (int k = 0; k < HH; ++k) acc += (float)h1[n * HH + k] * Wfc[k * HOR + o];
  out[tid] = acc;
}

extern "C" void kernel_launch(void* const* d_in, const int* in_sizes, int n_in,
                              void* d_out, int out_size, void* d_ws, size_t ws_size,
                              hipStream_t stream) {
  (void)in_sizes; (void)n_in; (void)out_size; (void)ws_size;
  const float* feats = (const float*)d_in[0];
  // d_in[1] = Wl1 (dead), d_in[4] = Wl2 (dead)
  const float* Wr1  = (const float*)d_in[2];
  const float* b1   = (const float*)d_in[3];
  const float* Wr2  = (const float*)d_in[5];
  const float* b2   = (const float*)d_in[6];
  const float* Wih0 = (const float*)d_in[7];
  const float* Whh0 = (const float*)d_in[8];
  const float* bih0 = (const float*)d_in[9];
  const float* bhh0 = (const float*)d_in[10];
  const float* Wih1 = (const float*)d_in[11];
  const float* Whh1 = (const float*)d_in[12];
  const float* bih1 = (const float*)d_in[13];
  const float* bhh1 = (const float*)d_in[14];
  const float* Wfc  = (const float*)d_in[15];
  const float* bfc  = (const float*)d_in[16];

  // Workspace layout
  char* ws = (char*)d_ws;
  const size_t XS_BYTES = (size_t)TT * BD * HH * sizeof(_Float16);   // 2 MB
  const size_t GI_BYTES = (size_t)TT * BD * G3 * sizeof(float);      // 12.6 MB
  _Float16* xs  = (_Float16*)ws;
  float*    Gi  = (float*)(ws + XS_BYTES);                           // reused per layer
  _Float16* ys0 = (_Float16*)(ws + XS_BYTES + GI_BYTES);
  _Float16* ys1 = (_Float16*)(ws + XS_BYTES + GI_BYTES + XS_BYTES);

  const size_t smem_seq = 128 * 64 * 2 + 128 * 128 * 2 + 4 * 16 * 128 * 2; // 64 KB
  const size_t smem_gi  = (size_t)G3 * HH * 2;                             // 96 KB
  const size_t smem_gru = (size_t)G3 * HH * 2 + (size_t)BD * HH * 2;       // 128 KB

  seq_kernel<<<128, 128, smem_seq, stream>>>(feats, Wr1, b1, Wr2, b2, xs);
  gi_kernel <<<512, 256, smem_gi,  stream>>>(xs,  Wih0, bih0, Gi);
  gru_kernel<<<1, 1024,  smem_gru, stream>>>(Gi,  Whh0, bhh0, ys0);
  gi_kernel <<<512, 256, smem_gi,  stream>>>(ys0, Wih1, bih1, Gi);
  gru_kernel<<<1, 1024,  smem_gru, stream>>>(Gi,  Whh1, bhh1, ys1);
  fc_kernel <<<(BD * HOR + 255) / 256, 256, 0, stream>>>(
      ys1 + (size_t)(TT - 1) * BD * HH, Wfc, bfc, (float*)d_out);
}
